// FingerprintDescriptorGNN_13469017441128
// MI455X (gfx1250) — compile-verified, matrix operand order checked by reading, not run on hardware
//
#include <hip/hip_runtime.h>
#include <stdint.h>

// ---------------- problem constants (fixed by reference) ----------------
#define N_NODES 50000
#define N_EDGES 600000
#define N_GRAPH 512
#define HDIM    128
#define OUTDIM  256
#define BN_EPS  1e-5f

// ---------------- CDNA5 WMMA types ----------------
typedef __bf16 v16bf __attribute__((ext_vector_type(16)));
typedef float  v8f   __attribute__((ext_vector_type(8)));

static __device__ __forceinline__ __bf16 f2bf(float f) {
  union { float f; uint32_t u; } x; x.f = f;
  uint32_t r = x.u + 0x7FFFu + ((x.u >> 16) & 1u);   // round-to-nearest-even
  uint16_t h = (uint16_t)(r >> 16);
  return __builtin_bit_cast(__bf16, h);
}

// ---------------- utility kernels ----------------
__global__ void k_zero(float* __restrict__ p, int n) {
  int i = blockIdx.x * blockDim.x + threadIdx.x;
  if (i < n) p[i] = 0.f;
}

// Pack a row-major [Kin,128] f32 weight into bf16 B-fragment order:
// out[((kt*128 + n)*2 + half)*16 + j] = W[kt*32 + half*16 + j][n]  (0 if k>=Kin)
// so each lane's v16bf B fragment is one contiguous, 32B-aligned load.
__global__ void k_pack(const float* __restrict__ W, __bf16* __restrict__ out,
                       int Kin, int KT) {
  int i = blockIdx.x * blockDim.x + threadIdx.x;
  int total = KT * HDIM * 32;
  if (i >= total) return;
  int j    = i & 15;
  int half = (i >> 4) & 1;
  int n    = (i >> 5) & (HDIM - 1);
  int kt   = i >> 12;                       // HDIM*32 == 4096
  int kk   = kt * 32 + half * 16 + j;
  out[i] = f2bf(kk < Kin ? W[(long)kk * HDIM + n] : 0.f);
}

__global__ void k_deg(const int* __restrict__ dst, float* __restrict__ deg) {
  int i = blockIdx.x * blockDim.x + threadIdx.x;
  if (i < N_EDGES) atomicAdd(&deg[dst[i]], 1.f);
}

__global__ void k_rdeg(float* __restrict__ deg) {
  int i = blockIdx.x * blockDim.x + threadIdx.x;
  if (i < N_NODES) deg[i] = 1.f / fmaxf(deg[i], 1.f);
}

// residual for layer 0: res = x @ Wr0 + br0   (N x 4 @ 4 x 128, trivial)
__global__ void k_res0(const float* __restrict__ x, const float* __restrict__ Wr,
                       const float* __restrict__ br, float* __restrict__ res) {
  int i = blockIdx.x * blockDim.x + threadIdx.x;
  if (i >= N_NODES * HDIM) return;
  int n = i >> 7, c = i & 127;
  float a = br[c];
#pragma unroll
  for (int k = 0; k < 4; ++k) a += x[n * 4 + k] * Wr[k * HDIM + c];
  res[i] = a;
}

// ---------------- edge message GEMM + scatter (WMMA) ----------------
// Block: 256 threads = 8 wave32. 32 edges x 128 output cols per block.
// A tile (32 x KT*32 bf16) staged in LDS: [h[src] | edge_attr | 0-pad].
// Each wave owns a 16-column stripe and TWO 16-row accumulators that share
// one B fragment per k-tile (halves weight traffic, doubles WMMA density).
template <int KT, int HK>
__global__ __launch_bounds__(256)
void k_edge_msg(const float* __restrict__ h, const float* __restrict__ ea,
                const int* __restrict__ src, const int* __restrict__ dst,
                const __bf16* __restrict__ Bpack, const float* __restrict__ be,
                float* __restrict__ agg) {
  constexpr int KP = KT * 32;        // padded K
  constexpr int AS = KP + 16;        // LDS row stride (bf16): 32B-aligned, conflict-spread
  __shared__ __bf16 Alds[32 * AS];
  __shared__ int dstL[32];

  int tid  = threadIdx.x;
  int row  = tid >> 3;               // edge slot 0..31 (8 threads per row)
  int c0   = tid & 7;
  int e    = blockIdx.x * 32 + row;
  int s    = -1;
  if (e < N_EDGES) {
    s = src[e];
    if (c0 == 0) dstL[row] = dst[e];
  } else if (c0 == 0) {
    dstL[row] = -1;
  }
  for (int c = c0; c < KP; c += 8) {
    float v = 0.f;
    if (s >= 0) {
      if (c < HK)          v = h[(long)s * HK + c];
      else if (c < HK + 2) v = ea[(long)e * 2 + (c - HK)];
    }
    Alds[row * AS + c] = f2bf(v);
  }
  __syncthreads();

  int lane  = tid & 31;              // wave32
  int wv    = tid >> 5;
  int lrow  = lane & 15;
  int lhalf = lane >> 4;
  int n0    = wv * 16;               // this wave's output-column base

  v8f acc0 = {};
  v8f acc1 = {};
#pragma unroll
  for (int kt = 0; kt < KT; ++kt) {
    // B frag: lane holds col N=n0+lrow, K = kt*32 + lhalf*16 + [0..15]
    v16bf b  = *(const v16bf*)(Bpack + (((long)kt * HDIM + n0 + lrow) * 2 + lhalf) * 16);
    // A frags: lane holds rows M=lrow and M=16+lrow, same K range (contiguous)
    v16bf a0 = *(const v16bf*)(Alds + lrow * AS        + kt * 32 + lhalf * 16);
    v16bf a1 = *(const v16bf*)(Alds + (16 + lrow) * AS + kt * 32 + lhalf * 16);
    acc0 = __builtin_amdgcn_wmma_f32_16x16x32_bf16(false, a0, false, b,
                                                   (short)0, acc0, false, false);
    acc1 = __builtin_amdgcn_wmma_f32_16x16x32_bf16(false, a1, false, b,
                                                   (short)0, acc1, false, false);
  }

  // C/D layout: lane = col, VGPR r = row (lanes 16-31 -> rows 8..15)
  int   col  = n0 + lrow;
  float bias = be[col];
#pragma unroll
  for (int r = 0; r < 8; ++r) {
    int m = lhalf * 8 + r;
    int d0 = dstL[m];
    if (d0 >= 0) atomicAdd(&agg[(long)d0 * HDIM + col], acc0[r] + bias);
    int d1 = dstL[16 + m];
    if (d1 >= 0) atomicAdd(&agg[(long)d1 * HDIM + col], acc1[r] + bias);
  }
}

// ---------------- node update GEMM (WMMA) + BN + residual + ReLU ----------------
// 32 nodes x 128 cols per block; two accumulators per wave share each B frag.
__global__ __launch_bounds__(256)
void k_node_update(const float* __restrict__ agg, const float* __restrict__ rdeg,
                   const __bf16* __restrict__ Bpack, const float* __restrict__ bno,
                   const float* __restrict__ gm, const float* __restrict__ bt,
                   const float* __restrict__ mn, const float* __restrict__ vr,
                   const float* __restrict__ res, float* __restrict__ hout) {
  constexpr int AS = HDIM + 16;
  __shared__ __bf16 Alds[32 * AS];

  int tid = threadIdx.x;
  int row = tid >> 3, c0 = tid & 7;
  int n   = blockIdx.x * 32 + row;
  float sc = (n < N_NODES) ? rdeg[n] : 0.f;
  for (int c = c0; c < HDIM; c += 8) {
    float v = (n < N_NODES) ? agg[(long)n * HDIM + c] * sc : 0.f;
    Alds[row * AS + c] = f2bf(v);
  }
  __syncthreads();

  int lane = tid & 31, wv = tid >> 5;
  int lrow = lane & 15, lhalf = lane >> 4;
  int n0 = wv * 16;

  v8f acc0 = {};
  v8f acc1 = {};
#pragma unroll
  for (int kt = 0; kt < 4; ++kt) {
    v16bf b  = *(const v16bf*)(Bpack + (((long)kt * HDIM + n0 + lrow) * 2 + lhalf) * 16);
    v16bf a0 = *(const v16bf*)(Alds + lrow * AS        + kt * 32 + lhalf * 16);
    v16bf a1 = *(const v16bf*)(Alds + (16 + lrow) * AS + kt * 32 + lhalf * 16);
    acc0 = __builtin_amdgcn_wmma_f32_16x16x32_bf16(false, a0, false, b,
                                                   (short)0, acc0, false, false);
    acc1 = __builtin_amdgcn_wmma_f32_16x16x32_bf16(false, a1, false, b,
                                                   (short)0, acc1, false, false);
  }

  int   col   = n0 + lrow;
  float bias  = bno[col];
  float scale = gm[col] * rsqrtf(vr[col] + BN_EPS);
  float shift = bt[col] - mn[col] * scale;
#pragma unroll
  for (int r = 0; r < 8; ++r) {
    int m   = lhalf * 8 + r;
    int nn0 = blockIdx.x * 32 + m;
    if (nn0 < N_NODES) {
      float val = (acc0[r] + bias) * scale + shift + res[(long)nn0 * HDIM + col];
      hout[(long)nn0 * HDIM + col] = fmaxf(val, 0.f);
    }
    int nn1 = nn0 + 16;
    if (nn1 < N_NODES) {
      float val = (acc1[r] + bias) * scale + shift + res[(long)nn1 * HDIM + col];
      hout[(long)nn1 * HDIM + col] = fmaxf(val, 0.f);
    }
  }
}

// ---------------- pooling + head ----------------
__global__ void k_pool(const float* __restrict__ h, const int* __restrict__ batch,
                       float* __restrict__ sum, float* __restrict__ cnt,
                       int* __restrict__ mx) {
  int i = blockIdx.x * blockDim.x + threadIdx.x;
  if (i >= N_NODES * HDIM) return;
  int n = i >> 7, c = i & 127;
  int g = batch[n];
  float hv = h[i];
  atomicAdd(&sum[(long)g * HDIM + c], hv);
  atomicMax(&mx[(long)g * HDIM + c], __float_as_int(hv)); // hv >= 0 after ReLU
  if (c == 0) atomicAdd(&cnt[g], 1.f);
}

__global__ void k_head1(const float* __restrict__ sum, const float* __restrict__ cnt,
                        const int* __restrict__ mx, const float* __restrict__ W1,
                        const float* __restrict__ b1, const float* __restrict__ g1,
                        const float* __restrict__ bt1, const float* __restrict__ m1,
                        const float* __restrict__ v1, float* __restrict__ z1) {
  int i = blockIdx.x * blockDim.x + threadIdx.x;
  if (i >= N_GRAPH * HDIM) return;
  int g = i >> 7, c = i & 127;
  float acc  = b1[c];
  float invc = 1.f / fmaxf(cnt[g], 1.f);
  for (int k = 0; k < HDIM; ++k)
    acc += (sum[(long)g * HDIM + k] * invc) * W1[k * HDIM + c];
  for (int k = 0; k < HDIM; ++k)
    acc += __int_as_float(mx[(long)g * HDIM + k]) * W1[(HDIM + k) * HDIM + c];
  float val = (acc - m1[c]) * (g1[c] * rsqrtf(v1[c] + BN_EPS)) + bt1[c];
  z1[i] = fmaxf(val, 0.f);
}

__global__ void k_head2(const float* __restrict__ z1, const float* __restrict__ W2,
                        const float* __restrict__ b2, float* __restrict__ out) {
  int i = blockIdx.x * blockDim.x + threadIdx.x;
  if (i >= N_GRAPH * OUTDIM) return;
  int g = i / OUTDIM, o = i % OUTDIM;
  float acc = b2[o];
  for (int k = 0; k < HDIM; ++k) acc += z1[(long)g * HDIM + k] * W2[k * OUTDIM + o];
  out[i] = acc;
}

__global__ void k_norm(float* __restrict__ out) {
  __shared__ float red[OUTDIM];
  int g = blockIdx.x, t = threadIdx.x;
  float v = out[(long)g * OUTDIM + t];
  red[t] = v * v;
  __syncthreads();
  for (int s = OUTDIM / 2; s > 0; s >>= 1) {
    if (t < s) red[t] += red[t + s];
    __syncthreads();
  }
  float nrm = fmaxf(sqrtf(red[0]), 1e-12f);
  out[(long)g * OUTDIM + t] = v / nrm;
}

// ---------------- host orchestration ----------------
extern "C" void kernel_launch(void* const* d_in, const int* in_sizes, int n_in,
                              void* d_out, int out_size, void* d_ws, size_t ws_size,
                              hipStream_t stream) {
  (void)in_sizes; (void)n_in; (void)out_size; (void)ws_size;
  const float* x     = (const float*)d_in[0];
  const float* ea    = (const float*)d_in[1];
  const float* We0   = (const float*)d_in[2];
  const float* be0   = (const float*)d_in[3];
  const float* Wn0   = (const float*)d_in[4];
  const float* bno0  = (const float*)d_in[5];
  const float* g0    = (const float*)d_in[6];
  const float* bt0   = (const float*)d_in[7];
  const float* m0    = (const float*)d_in[8];
  const float* v0    = (const float*)d_in[9];
  const float* Wr0   = (const float*)d_in[10];
  const float* br0   = (const float*)d_in[11];
  const float* We_s  = (const float*)d_in[12];
  const float* be_s  = (const float*)d_in[13];
  const float* Wn_s  = (const float*)d_in[14];
  const float* bno_s = (const float*)d_in[15];
  const float* g_s   = (const float*)d_in[16];
  const float* bt_s  = (const float*)d_in[17];
  const float* m_s   = (const float*)d_in[18];
  const float* v_s   = (const float*)d_in[19];
  const float* W1    = (const float*)d_in[20];
  const float* b1    = (const float*)d_in[21];
  const float* g1    = (const float*)d_in[22];
  const float* bt1   = (const float*)d_in[23];
  const float* m1    = (const float*)d_in[24];
  const float* v1    = (const float*)d_in[25];
  const float* W2    = (const float*)d_in[26];
  const float* b2    = (const float*)d_in[27];
  const int*   ei    = (const int*)d_in[28];
  const int*   batch = (const int*)d_in[29];
  const int* src = ei;
  const int* dst = ei + N_EDGES;

  // bump allocator over workspace
  char* base = (char*)d_ws;
  size_t off = 0;
  auto alloc = [&](size_t bytes) -> char* {
    char* p = base + off;
    off += (bytes + 255) & ~(size_t)255;
    return p;
  };
  float* deg  = (float*)alloc((size_t)N_NODES * 4);
  float* agg  = (float*)alloc((size_t)N_NODES * HDIM * 4);
  float* bufA = (float*)alloc((size_t)N_NODES * HDIM * 4);
  float* bufB = (float*)alloc((size_t)N_NODES * HDIM * 4);
  __bf16* pWe0 = (__bf16*)alloc((size_t)1 * HDIM * 32 * 2);
  __bf16* pWn0 = (__bf16*)alloc((size_t)4 * HDIM * 32 * 2);
  __bf16* pWe[7]; __bf16* pWn[7];
  for (int i = 0; i < 7; ++i) {
    pWe[i] = (__bf16*)alloc((size_t)5 * HDIM * 32 * 2);
    pWn[i] = (__bf16*)alloc((size_t)4 * HDIM * 32 * 2);
  }
  float* psum = (float*)alloc((size_t)N_GRAPH * HDIM * 4);
  float* pcnt = (float*)alloc((size_t)N_GRAPH * 4);
  int*   pmax = (int*)alloc((size_t)N_GRAPH * HDIM * 4);
  float* z1   = (float*)alloc((size_t)N_GRAPH * HDIM * 4);

  // pack all weights into WMMA B-fragment order (deterministic, per call)
  k_pack<<<(1 * HDIM * 32 + 255) / 256, 256, 0, stream>>>(We0, pWe0, 6, 1);
  k_pack<<<(4 * HDIM * 32 + 255) / 256, 256, 0, stream>>>(Wn0, pWn0, 128, 4);
  for (int i = 0; i < 7; ++i) {
    k_pack<<<(5 * HDIM * 32 + 255) / 256, 256, 0, stream>>>(
        We_s + (size_t)i * 130 * HDIM, pWe[i], 130, 5);
    k_pack<<<(4 * HDIM * 32 + 255) / 256, 256, 0, stream>>>(
        Wn_s + (size_t)i * HDIM * HDIM, pWn[i], 128, 4);
  }

  // reciprocal in-degree
  k_zero<<<(N_NODES + 255) / 256, 256, 0, stream>>>(deg, N_NODES);
  k_deg<<<(N_EDGES + 255) / 256, 256, 0, stream>>>(dst, deg);
  k_rdeg<<<(N_NODES + 255) / 256, 256, 0, stream>>>(deg);

  // layer-0 residual projection
  k_res0<<<(N_NODES * HDIM + 255) / 256, 256, 0, stream>>>(x, Wr0, br0, bufA);

  const int NH      = N_NODES * HDIM;
  const int eblocks = (N_EDGES + 31) / 32;
  const int nblocks = (N_NODES + 31) / 32;

  // layer 0: 4-feature input, K padded 6->32
  k_zero<<<(NH + 255) / 256, 256, 0, stream>>>(agg, NH);
  k_edge_msg<1, 4><<<eblocks, 256, 0, stream>>>(x, ea, src, dst, pWe0, be0, agg);
  k_node_update<<<nblocks, 256, 0, stream>>>(agg, deg, pWn0, bno0, g0, bt0, m0, v0,
                                             bufA, bufB);
  float* cur = bufB;
  float* nxt = bufA;
  // layers 1..7: K padded 130->160
  for (int i = 0; i < 7; ++i) {
    k_zero<<<(NH + 255) / 256, 256, 0, stream>>>(agg, NH);
    k_edge_msg<5, 128><<<eblocks, 256, 0, stream>>>(cur, ea, src, dst, pWe[i],
                                                    be_s + i * HDIM, agg);
    k_node_update<<<nblocks, 256, 0, stream>>>(
        agg, deg, pWn[i], bno_s + i * HDIM, g_s + i * HDIM, bt_s + i * HDIM,
        m_s + i * HDIM, v_s + i * HDIM, cur, nxt);
    float* t = cur; cur = nxt; nxt = t;
  }

  // pooling + head + normalize
  k_zero<<<(N_GRAPH * HDIM + 255) / 256, 256, 0, stream>>>(psum, N_GRAPH * HDIM);
  k_zero<<<(N_GRAPH + 255) / 256, 256, 0, stream>>>(pcnt, N_GRAPH);
  k_zero<<<(N_GRAPH * HDIM + 255) / 256, 256, 0, stream>>>((float*)pmax, N_GRAPH * HDIM);
  k_pool<<<(NH + 255) / 256, 256, 0, stream>>>(cur, batch, psum, pcnt, pmax);
  k_head1<<<(N_GRAPH * HDIM + 255) / 256, 256, 0, stream>>>(psum, pcnt, pmax, W1, b1,
                                                            g1, bt1, m1, v1, z1);
  k_head2<<<(N_GRAPH * OUTDIM + 255) / 256, 256, 0, stream>>>(z1, W2, b2,
                                                              (float*)d_out);
  k_norm<<<N_GRAPH, OUTDIM, 0, stream>>>((float*)d_out);
}
